// MACE_36945308680468
// MI455X (gfx1250) — matrix operand order repaired and checked by reading, not compiled, not run on hardware
//
#include <hip/hip_runtime.h>
#include <hip/hip_bf16.h>
#include <math.h>

// ---------------------------------------------------------------------------
// MACE forward for MI455X (gfx1250): wave32 + bf16 WMMA 16x16x32
// N=4096 nodes, E=65536 edges, C=128 channels, Z=10, G=16 graphs
// Weights are pre-transposed+converted to bf16 so B-fragments are two
// contiguous 128-bit loads; A-fragments are two ds_load_b128 from LDS.
// ---------------------------------------------------------------------------

#define kN 4096
#define kE 65536
#define kC 128
#define kZ 10
#define kG 16
#define kNB 8
#define INV_AVG 0.0625f

typedef __attribute__((ext_vector_type(16))) __bf16 v16bf;
typedef __attribute__((ext_vector_type(8)))  float  v8f;

__device__ inline unsigned short f2bf_bits(float f) {
    unsigned u = __float_as_uint(f);
    unsigned r = u + 0x7FFFu + ((u >> 16) & 1u);   // round-to-nearest-even
    return (unsigned short)(r >> 16);
}
__device__ inline unsigned pack2(float a, float b) {
    return (unsigned)f2bf_bits(a) | ((unsigned)f2bf_bits(b) << 16);
}
__device__ inline float silu(float x) { return x / (1.f + __expf(-x)); }
__device__ inline int lofk(int k) { return (k == 0) ? 0 : (k < 4 ? 1 : (k < 9 ? 2 : 3)); }
__device__ inline void atomAddF(float* p, float v) { unsafeAtomicAdd(p, v); }

__device__ inline v8f wmma_bf16(v16bf a, v16bf b, v8f c) {
    return __builtin_amdgcn_wmma_f32_16x16x32_bf16(false, a, false, b, (short)0, c, false, false);
}

// A-matrix fragment 16x32 bf16 from row-major tile ('stride' elems/row).
// Per ISA 7.12.2: lane half -> K+8, vgprs 4-7 -> K+16. Per-lane data is two
// contiguous 8-element (16B) runs -> two 128-bit loads.
__device__ inline v16bf load_a_frag(const unsigned short* base, int stride, int lane,
                                    int row0, int k0) {
    int row = row0 + (lane & 15);
    const unsigned short* p = base + (size_t)row * stride + k0 + (lane >> 4) * 8;
    union { uint4 q[2]; v16bf v; } u;
    u.q[0] = *(const uint4*)(p);
    u.q[1] = *(const uint4*)(p + 16);
    return u.v;
}

// B-matrix fragment 32x16 bf16 from a TRANSPOSED weight [n][k] ('stride'=K).
// Per-lane data is K = half*8 + 0..7 and 16 + half*8 + 0..7 at fixed column n
// (same order as A) -> two 16-byte runs -> two 128-bit loads.
__device__ inline v16bf load_b_fragT(const unsigned short* baseT, int stride, int lane,
                                     int k0, int col0) {
    int n = col0 + (lane & 15);
    const unsigned short* p = baseT + (size_t)n * stride + k0 + (lane >> 4) * 8;
    union { uint4 q[2]; v16bf v; } u;
    u.q[0] = *(const uint4*)(p);
    u.q[1] = *(const uint4*)(p + 16);
    return u.v;
}

// Store a 16x16 f32 D-fragment into bf16 LDS with SiLU applied.
__device__ inline void store_silu_bf16(unsigned short* dst, int stride, int colBase,
                                       int lane, v8f acc) {
    int half = lane >> 4, col = colBase + (lane & 15);
#pragma unroll
    for (int r = 0; r < 8; r++) {
        int row = r + 8 * half;
        dst[row * stride + col] = f2bf_bits(silu(acc[r]));
    }
}

// ---------------------------------------------------------------------------
// Utility kernels
// ---------------------------------------------------------------------------
__global__ void MACE_zero(float* p, int count) {
    int g = blockIdx.x * blockDim.x + threadIdx.x;
    if (g < count) p[g] = 0.f;
}

// Transpose+convert weight: in f32 [K][N] -> out bf16 [N][Kpad] (zero-padded K)
__global__ void MACE_twt(const float* __restrict__ in, unsigned short* __restrict__ out,
                         int K, int N, int Kpad) {
    int idx = blockIdx.x * blockDim.x + threadIdx.x;
    if (idx >= N * Kpad) return;
    int n = idx / Kpad, k = idx % Kpad;
    out[idx] = (k < K) ? f2bf_bits(in[(size_t)k * N + n]) : (unsigned short)0;
}

__global__ void MACE_node_init(const float* __restrict__ node_attrs,
                               const float* __restrict__ atomic_energies,
                               const float* __restrict__ W_emb,
                               int* __restrict__ species,
                               float* __restrict__ e0,
                               float* __restrict__ h) {
    int g = blockIdx.x * blockDim.x + threadIdx.x;
    if (g >= kN * kC) return;
    int n = g >> 7, c = g & 127;
    int sp = 0;
#pragma unroll
    for (int j = 0; j < kZ; j++)
        if (node_attrs[n * kZ + j] > 0.5f) sp = j;
    h[g] = W_emb[sp * kC + c];
    if (c == 0) { species[n] = sp; e0[n] = atomic_energies[sp]; }
}

__global__ void MACE_edge_geom(const float* __restrict__ pos,
                               const float* __restrict__ shifts,
                               const int* __restrict__ edge_index,
                               float* __restrict__ Y,
                               float* __restrict__ B) {
    int e = blockIdx.x * blockDim.x + threadIdx.x;
    if (e >= kE) return;
    int snd = edge_index[e], rcv = edge_index[kE + e];
    float vx = pos[rcv * 3 + 0] - pos[snd * 3 + 0] + shifts[e * 3 + 0];
    float vy = pos[rcv * 3 + 1] - pos[snd * 3 + 1] + shifts[e * 3 + 1];
    float vz = pos[rcv * 3 + 2] - pos[snd * 3 + 2] + shifts[e * 3 + 2];
    float r = sqrtf(vx * vx + vy * vy + vz * vz);
    float rinv = 1.f / fmaxf(r, 1e-12f);
    float x = vx * rinv, y = vy * rinv, z = vz * rinv;
    const float s3 = 1.73205081f, s5 = 2.23606798f, s15 = 3.87298335f;
    const float s7 = 2.64575131f, s42 = 6.48074069f, s70 = 8.36660027f, s105 = 10.24695077f;
    float x2 = x * x, y2 = y * y, z2 = z * z;
    float* Ye = Y + (size_t)e * 16;
    Ye[0]  = 1.f;
    Ye[1]  = s3 * x;  Ye[2] = s3 * y;  Ye[3] = s3 * z;
    Ye[4]  = s15 * x * y;
    Ye[5]  = s15 * y * z;
    Ye[6]  = 0.5f * s5 * (3.f * z2 - 1.f);
    Ye[7]  = s15 * x * z;
    Ye[8]  = 0.5f * s15 * (x2 - y2);
    Ye[9]  = 0.25f * s70 * y * (3.f * x2 - y2);
    Ye[10] = s105 * x * y * z;
    Ye[11] = 0.25f * s42 * y * (5.f * z2 - 1.f);
    Ye[12] = 0.5f * s7 * z * (5.f * z2 - 3.f);
    Ye[13] = 0.25f * s42 * x * (5.f * z2 - 1.f);
    Ye[14] = 0.5f * s105 * z * (x2 - y2);
    Ye[15] = 0.25f * s70 * x * (x2 - 3.f * y2);
    float u = r * 0.2f;
    float env = 0.f;
    if (u < 1.f) {
        float u2 = u * u, u4 = u2 * u2, u5 = u4 * u;
        env = 1.f - 21.f * u5 + 35.f * u5 * u - 15.f * u5 * u2;
    }
    float c0 = 0.632455532f * rinv * env;
#pragma unroll
    for (int nb = 1; nb <= kNB; nb++)
        B[(size_t)e * kNB + nb - 1] = c0 * __sinf(3.14159265358979f * nb * u);
}

// ---------------------------------------------------------------------------
// Batched GEMM (M x 128 x 128): Out[z] = A[z] @ Wt[map(z)]^T-layout (+ Add).
// Block = 4 waves, tile 64(M) x 64(N). A staged once (64x128 bf16 in LDS),
// B-fragments straight from pre-transposed bf16 weights in global (L2-hot).
// ---------------------------------------------------------------------------
__global__ __launch_bounds__(128)
void MACE_gemm(const float* __restrict__ A, int aRow, int aBatch,
               const unsigned short* __restrict__ Wt, int wBatch, int wMap,
               const float* __restrict__ Add,
               float* __restrict__ Out, int oRow, int oBatch) {
    __shared__ __align__(16) unsigned short sA[64 * 128];
    int tid = threadIdx.x, lane = tid & 31, wave = tid >> 5;
    int mBase = blockIdx.x * 64, nBase = blockIdx.y * 64, z = blockIdx.z;
    const float* Ap = A + (size_t)z * aBatch;
    const unsigned short* Wp = Wt + (size_t)((wMap == 1) ? lofk(z) : z) * wBatch;

    // stage A tile 64x128 (f32 -> packed bf16 pairs)
    for (int i = tid; i < 64 * 64; i += 128) {
        int row = i >> 6, kp = (i & 63) * 2;
        const float* src = Ap + (size_t)(mBase + row) * aRow + kp;
        ((unsigned*)sA)[i] = pack2(src[0], src[1]);
    }
    __syncthreads();

    v8f acc[4];
#pragma unroll
    for (int t = 0; t < 4; t++) acc[t] = {};
#pragma unroll
    for (int kb = 0; kb < 128; kb += 32) {
        v16bf a = load_a_frag(sA, 128, lane, wave * 16, kb);
#pragma unroll
        for (int t = 0; t < 4; t++) {
            v16bf b = load_b_fragT(Wp, 128, lane, kb, nBase + t * 16);
            acc[t] = wmma_bf16(a, b, acc[t]);
        }
    }

    int half = lane >> 4, coll = lane & 15;
#pragma unroll
    for (int t = 0; t < 4; t++) {
#pragma unroll
        for (int r = 0; r < 8; r++) {
            int row = mBase + wave * 16 + r + 8 * half;
            int col = nBase + t * 16 + coll;
            size_t oi = (size_t)z * oBatch + (size_t)row * oRow + col;
            float v = acc[t][r];
            if (Add) v += Add[oi];
            Out[oi] = v;
        }
    }
}

// ---------------------------------------------------------------------------
// Fused radial-MLP + message-scatter kernel.
// 128 threads (4 waves) own 16 edges. MLP 8->64->64->64->ncols via bf16 WMMA;
// activations relayouted through LDS; weights read directly from global
// pre-transposed bf16. Messages scattered into A[rcv,16,128] with f32 atomics.
// ncols = 512 (layer 1) or 640 (layer 2; block l==4 is the vec1-dot term).
// ---------------------------------------------------------------------------
__global__ __launch_bounds__(128)
void MACE_edge_mlp_msg(const float* __restrict__ Bfeat,   // [E,8]
                       const float* __restrict__ Y,       // [E,16]
                       const int* __restrict__ edge_index,// [2,E]
                       const unsigned short* __restrict__ W1t, // [64][32] bf16
                       const unsigned short* __restrict__ W2t, // [64][64] bf16
                       const unsigned short* __restrict__ W3t, // [64][64] bf16
                       const unsigned short* __restrict__ W4t, // [ncols][64] bf16
                       const float* __restrict__ hs,      // [N,128]
                       const float* __restrict__ vec1,    // [N,3,128] or null
                       float* __restrict__ Aout,          // [N,16,128]
                       int ncols) {
    __shared__ __align__(16) unsigned short sIn[16 * 32];
    __shared__ __align__(16) unsigned short sH[2][16 * 64];
    __shared__ float sY[16 * 16];
    __shared__ float sHs[16 * 128];
    __shared__ int sSnd[16], sRcv[16];

    int tid = threadIdx.x, lane = tid & 31, wave = tid >> 5;
    int eBase = blockIdx.x * 16;

    // ---- stage B tile (K padded 8->32, packed pairs), Y tile, indices
    for (int i = tid; i < 16 * 16; i += 128) {
        int row = i >> 4, kp = (i & 15) * 2;
        float v0 = (kp < kNB)     ? Bfeat[(size_t)(eBase + row) * kNB + kp]     : 0.f;
        float v1 = (kp + 1 < kNB) ? Bfeat[(size_t)(eBase + row) * kNB + kp + 1] : 0.f;
        ((unsigned*)sIn)[i] = pack2(v0, v1);
    }
    for (int i = tid; i < 256; i += 128) sY[i] = Y[(size_t)eBase * 16 + i];
    if (tid < 16) {
        sSnd[tid] = edge_index[eBase + tid];
        sRcv[tid] = edge_index[kE + eBase + tid];
    }
    __syncthreads();

    // ---- stage hs rows for the 16 edges (used by the message scatter)
    for (int i = tid; i < 16 * 128; i += 128)
        sHs[i] = hs[(size_t)sSnd[i >> 7] * kC + (i & 127)];

    // ---- H1 = silu(B @ W1)
    {
        v16bf a = load_a_frag(sIn, 32, lane, 0, 0);
        v16bf b = load_b_fragT(W1t, 32, lane, 0, wave * 16);
        v8f acc = {};
        acc = wmma_bf16(a, b, acc);
        store_silu_bf16(sH[0], 64, wave * 16, lane, acc);
    }
    __syncthreads();

    // ---- H2 = silu(H1 @ W2)
    {
        v8f acc = {};
        v16bf a0 = load_a_frag(sH[0], 64, lane, 0, 0);
        v16bf a1 = load_a_frag(sH[0], 64, lane, 0, 32);
        acc = wmma_bf16(a0, load_b_fragT(W2t, 64, lane, 0,  wave * 16), acc);
        acc = wmma_bf16(a1, load_b_fragT(W2t, 64, lane, 32, wave * 16), acc);
        store_silu_bf16(sH[1], 64, wave * 16, lane, acc);
    }
    __syncthreads();

    // ---- H3 = silu(H2 @ W3)
    {
        v8f acc = {};
        v16bf a0 = load_a_frag(sH[1], 64, lane, 0, 0);
        v16bf a1 = load_a_frag(sH[1], 64, lane, 0, 32);
        acc = wmma_bf16(a0, load_b_fragT(W3t, 64, lane, 0,  wave * 16), acc);
        acc = wmma_bf16(a1, load_b_fragT(W3t, 64, lane, 32, wave * 16), acc);
        store_silu_bf16(sH[0], 64, wave * 16, lane, acc);
    }
    __syncthreads();

    // ---- R = H3 @ W4, fused with message scatter (R never materialized)
    int half = lane >> 4, coll = lane & 15;
    v16bf a0 = load_a_frag(sH[0], 64, lane, 0, 0);
    v16bf a1 = load_a_frag(sH[0], 64, lane, 0, 32);
    int nChunks = ncols >> 6;
    for (int cc = 0; cc < nChunks; cc++) {
        int colBase = cc * 64 + wave * 16;
        v8f acc = {};
        acc = wmma_bf16(a0, load_b_fragT(W4t, 64, lane, 0,  colBase), acc);
        acc = wmma_bf16(a1, load_b_fragT(W4t, 64, lane, 32, colBase), acc);
#pragma unroll
        for (int r = 0; r < 8; r++) {
            int row = r + 8 * half;           // edge index within tile
            float val = acc[r];
            int j = colBase + coll;
            int l = j >> 7, c = j & 127;
            int rcvBase = sRcv[row] * 2048;
            if (l < 4) {
                float mv = val * sHs[row * 128 + c] * INV_AVG;
                int k0 = (l == 0) ? 0 : (l == 1) ? 1 : (l == 2) ? 4 : 9;
                int k1 = (l == 0) ? 1 : (l == 1) ? 4 : (l == 2) ? 9 : 16;
                for (int k = k0; k < k1; k++)
                    atomAddF(&Aout[rcvBase + k * kC + c], sY[row * 16 + k] * mv);
            } else if (vec1) {
                // layer-2 extra: m2[:,0,:] += R2[:,4,:] * dot(vec1[snd], Y[1:4])
                const float* vp = vec1 + (size_t)sSnd[row] * 384 + c;
                float dotv = vp[0]   * sY[row * 16 + 1]
                           + vp[128] * sY[row * 16 + 2]
                           + vp[256] * sY[row * 16 + 3];
                atomAddF(&Aout[rcvBase + c], val * dotv * INV_AVG);
            }
        }
    }
}

// ---------------------------------------------------------------------------
// Per-node invariant basis: s (scalar channel weights) and vec1 pre-mix
// ---------------------------------------------------------------------------
__global__ __launch_bounds__(128)
void MACE_basis(const float* __restrict__ Atil,    // [N,16,128]
                const int* __restrict__ species,
                const float* __restrict__ pw0sel,  // [Z,9,128]
                const float* __restrict__ pwv,     // [Z,3,128] or null
                float* __restrict__ sOut,          // [N,128]
                float* __restrict__ vpOut) {       // [N,3,128] or null
    int n = blockIdx.x, c = threadIdx.x;
    const float* Ar = Atil + (size_t)n * 2048 + c;
    float At[16];
#pragma unroll
    for (int k = 0; k < 16; k++) At[k] = Ar[k * kC];
    float A0 = At[0];
    float p0 = A0 * A0;
    float p1 = At[1] * At[1] + At[2] * At[2] + At[3] * At[3];
    float p2 = 0.f, p3 = 0.f;
#pragma unroll
    for (int k = 4; k < 9; k++) p2 += At[k] * At[k];
#pragma unroll
    for (int k = 9; k < 16; k++) p3 += At[k] * At[k];
    float bs[9] = { A0, p0, p1, p2, p3, A0 * p0, A0 * p1, A0 * p2, A0 * p3 };
    int sp = species[n];
    const float* pw = pw0sel + (size_t)sp * 9 * kC + c;
    float s = 0.f;
#pragma unroll
    for (int bi = 0; bi < 9; bi++) s += pw[bi * kC] * bs[bi];
    sOut[(size_t)n * kC + c] = s;
    if (pwv) {
        const float* pv = pwv + (size_t)sp * 384 + c;
        float wv = pv[0] + pv[128] * A0 + pv[256] * A0 * A0;
#pragma unroll
        for (int m = 0; m < 3; m++)
            vpOut[(size_t)n * 384 + m * kC + c] = At[1 + m] * wv;
    }
}

// sc[n,d] = scal1[n,:] @ W_sc[species[n],:,d]
__global__ __launch_bounds__(128)
void MACE_sc(const float* __restrict__ scal1, const int* __restrict__ species,
             const float* __restrict__ W_sc, float* __restrict__ out) {
    __shared__ float s[128];
    int n = blockIdx.x, d = threadIdx.x;
    s[d] = scal1[(size_t)n * kC + d];
    __syncthreads();
    int sp = species[n];
    const float* W = W_sc + (size_t)sp * kC * kC + d;
    float acc = 0.f;
    for (int c = 0; c < kC; c++) acc += s[c] * W[(size_t)c * kC];
    out[(size_t)n * kC + d] = acc;
}

// node_energy = e0 + scal1.r1 + silu(scal2@r2a).r2b ; atomic into out[batch[n]]
__global__ __launch_bounds__(128)
void MACE_energy(const float* __restrict__ scal1, const float* __restrict__ scal2,
                 const float* __restrict__ e0, const int* __restrict__ batch,
                 const float* __restrict__ r1, const float* __restrict__ r2a,
                 const float* __restrict__ r2b, float* __restrict__ out) {
    __shared__ float red[128];
    __shared__ float s2[128];
    __shared__ float t16[16];
    int n = blockIdx.x, c = threadIdx.x;
    s2[c] = scal2[(size_t)n * kC + c];
    red[c] = scal1[(size_t)n * kC + c] * r1[c];
    __syncthreads();
    for (int ofs = 64; ofs > 0; ofs >>= 1) {
        if (c < ofs) red[c] += red[c + ofs];
        __syncthreads();
    }
    if (c < 16) {
        float t = 0.f;
        for (int cc = 0; cc < kC; cc++) t += s2[cc] * r2a[cc * 16 + c];
        t16[c] = silu(t) * r2b[c];
    }
    __syncthreads();
    if (c == 0) {
        float e2 = 0.f;
#pragma unroll
        for (int j = 0; j < 16; j++) e2 += t16[j];
        atomAddF(&out[batch[n]], e0[n] + red[0] + e2);
    }
}

// ---------------------------------------------------------------------------
// Host orchestration
// ---------------------------------------------------------------------------
extern "C" void kernel_launch(void* const* d_in, const int* in_sizes, int n_in,
                              void* d_out, int out_size, void* d_ws, size_t ws_size,
                              hipStream_t stream) {
    const float* node_attrs      = (const float*)d_in[0];
    const float* positions       = (const float*)d_in[1];
    const float* shifts          = (const float*)d_in[2];
    const int*   edge_index      = (const int*)d_in[3];
    const int*   batch           = (const int*)d_in[4];
    const float* atomic_energies = (const float*)d_in[5];
    const float* W_emb           = (const float*)d_in[6];
    const float* rW1             = (const float*)d_in[7];
    const float* rW2             = (const float*)d_in[8];
    const float* rW3             = (const float*)d_in[9];
    const float* rW4a            = (const float*)d_in[10];
    const float* rW4b            = (const float*)d_in[11];
    const float* W_up            = (const float*)d_in[12];
    const float* W_msg           = (const float*)d_in[13];
    const float* W_sc            = (const float*)d_in[14];
    const float* pw0             = (const float*)d_in[15];
    const float* pwv             = (const float*)d_in[16];
    const float* Ws_mix          = (const float*)d_in[17];
    const float* Wv_mix          = (const float*)d_in[18];
    const float* r1              = (const float*)d_in[19];
    const float* r2a             = (const float*)d_in[20];
    const float* r2b             = (const float*)d_in[21];
    float* out = (float*)d_out;

    // ---- workspace carve-up (~93 MB)
    char* ws = (char*)d_ws;
    size_t off = 0;
    auto take = [&](size_t bytes) -> void* {
        void* p = ws + off;
        off = (off + bytes + 255) & ~(size_t)255;
        return p;
    };
    int*   species = (int*)  take((size_t)kN * 4);
    float* e0      = (float*)take((size_t)kN * 4);
    float* h       = (float*)take((size_t)kN * kC * 4);
    float* Yb      = (float*)take((size_t)kE * 16 * 4);
    float* Bb      = (float*)take((size_t)kE * kNB * 4);
    float* hsB     = (float*)take((size_t)kN * kC * 4);        // hs, then hs2
    float* Abuf    = (float*)take((size_t)kN * 2048 * 4);      // A, then A2
    float* Atil    = (float*)take((size_t)kN * 2048 * 4);      // Atil, then A2til
    float* sB      = (float*)take((size_t)kN * kC * 4);        // s, then s2
    float* scal1B  = (float*)take((size_t)kN * kC * 4);
    float* vpB     = (float*)take((size_t)kN * 3 * kC * 4);
    float* vec1B   = (float*)take((size_t)kN * 3 * kC * 4);
    float* scB     = (float*)take((size_t)kN * kC * 4);
    float* scal2B  = (float*)take((size_t)kN * kC * 4);
    // pre-transposed bf16 weights
    unsigned short* w1t    = (unsigned short*)take(2 * 64 * 32 * 2);
    unsigned short* w2t    = (unsigned short*)take(2 * 64 * 64 * 2);
    unsigned short* w3t    = (unsigned short*)take(2 * 64 * 64 * 2);
    unsigned short* w4at   = (unsigned short*)take((size_t)512 * 64 * 2);
    unsigned short* w4bt   = (unsigned short*)take((size_t)640 * 64 * 2);
    unsigned short* wupT   = (unsigned short*)take(2 * 16384 * 2);
    unsigned short* wmsgT  = (unsigned short*)take(8 * 16384 * 2);
    unsigned short* wsmixT = (unsigned short*)take(2 * 16384 * 2);
    unsigned short* wvmixT = (unsigned short*)take((size_t)16384 * 2);

    const int AC = kN * 2048;  // elements in A

    // ---- weight prep (bf16 transposed)
    for (int l = 0; l < 2; l++) {
        MACE_twt<<<(64 * 32 + 255) / 256, 256, 0, stream>>>(rW1 + l * kNB * 64, w1t + l * 64 * 32, 8, 64, 32);
        MACE_twt<<<(64 * 64 + 255) / 256, 256, 0, stream>>>(rW2 + l * 4096, w2t + l * 4096, 64, 64, 64);
        MACE_twt<<<(64 * 64 + 255) / 256, 256, 0, stream>>>(rW3 + l * 4096, w3t + l * 4096, 64, 64, 64);
        MACE_twt<<<(16384 + 255) / 256, 256, 0, stream>>>(W_up + l * 16384, wupT + l * 16384, 128, 128, 128);
        MACE_twt<<<(16384 + 255) / 256, 256, 0, stream>>>(Ws_mix + l * 16384, wsmixT + l * 16384, 128, 128, 128);
    }
    MACE_twt<<<(512 * 64 + 255) / 256, 256, 0, stream>>>(rW4a, w4at, 64, 512, 64);
    MACE_twt<<<(640 * 64 + 255) / 256, 256, 0, stream>>>(rW4b, w4bt, 64, 640, 64);
    for (int i = 0; i < 8; i++)
        MACE_twt<<<(16384 + 255) / 256, 256, 0, stream>>>(W_msg + i * 16384, wmsgT + i * 16384, 128, 128, 128);
    MACE_twt<<<(16384 + 255) / 256, 256, 0, stream>>>(Wv_mix, wvmixT, 128, 128, 128);

    // ---- phase 0: init
    MACE_zero<<<(AC + 255) / 256, 256, 0, stream>>>(Abuf, AC);
    MACE_zero<<<1, 32, 0, stream>>>(out, kG);
    MACE_node_init<<<(kN * kC + 255) / 256, 256, 0, stream>>>(
        node_attrs, atomic_energies, W_emb, species, e0, h);
    MACE_edge_geom<<<kE / 256, 256, 0, stream>>>(positions, shifts, edge_index, Yb, Bb);

    // ---- layer 1
    MACE_gemm<<<dim3(kN / 64, kC / 64, 1), 128, 0, stream>>>(
        h, kC, 0, wupT, 0, 0, nullptr, hsB, kC, 0);
    MACE_edge_mlp_msg<<<kE / 16, 128, 0, stream>>>(
        Bb, Yb, edge_index, w1t, w2t, w3t, w4at, hsB, nullptr, Abuf, 512);
    MACE_gemm<<<dim3(kN / 64, kC / 64, 16), 128, 0, stream>>>(
        Abuf, 2048, kC, wmsgT, 16384, 1, nullptr, Atil, 2048, kC);
    MACE_basis<<<kN, 128, 0, stream>>>(Atil, species, pw0, pwv, sB, vpB);
    MACE_zero<<<(AC + 255) / 256, 256, 0, stream>>>(Abuf, AC);   // recycle as A2
    MACE_gemm<<<dim3(kN / 64, kC / 64, 1), 128, 0, stream>>>(
        sB, kC, 0, wsmixT, 0, 0, nullptr, scal1B, kC, 0);
    MACE_gemm<<<dim3(kN / 64, kC / 64, 3), 128, 0, stream>>>(
        vpB, 384, kC, wvmixT, 0, 0, nullptr, vec1B, 384, kC);

    // ---- layer 2
    MACE_gemm<<<dim3(kN / 64, kC / 64, 1), 128, 0, stream>>>(
        scal1B, kC, 0, wupT + 16384, 0, 0, nullptr, hsB, kC, 0);
    MACE_edge_mlp_msg<<<kE / 16, 128, 0, stream>>>(
        Bb, Yb, edge_index, w1t + 64 * 32, w2t + 4096, w3t + 4096, w4bt,
        hsB, vec1B, Abuf, 640);
    MACE_gemm<<<dim3(kN / 64, kC / 64, 16), 128, 0, stream>>>(
        Abuf, 2048, kC, wmsgT + 4 * 16384, 16384, 1, nullptr, Atil, 2048, kC);
    MACE_basis<<<kN, 128, 0, stream>>>(Atil, species, pw0 + kZ * 9 * kC, nullptr, sB, nullptr);
    MACE_sc<<<kN, 128, 0, stream>>>(scal1B, species, W_sc, scB);
    MACE_gemm<<<dim3(kN / 64, kC / 64, 1), 128, 0, stream>>>(
        sB, kC, 0, wsmixT + 16384, 0, 0, scB, scal2B, kC, 0);

    // ---- readout
    MACE_energy<<<kN, 128, 0, stream>>>(scal1B, scal2B, e0, batch, r1, r2a, r2b, out);
}
